// ConfidenceAttention_9637906612660
// MI455X (gfx1250) — matrix operand order repaired and checked by reading, not compiled, not run on hardware
//
#include <hip/hip_runtime.h>
#include <hip/hip_bf16.h>
#include <stdint.h>

// ---------------------------------------------------------------------------
// ConfidenceAttention for MI455X (gfx1250)
//   - all matmuls via v_wmma_f32_16x16x32_bf16
//   - attention K/V tiles staged to LDS via global_load_async_to_lds_b128
//     (ASYNCcnt), V transfer overlapped behind QK^T + softmax
// B=256, N=256, HID=512, NH=8, HD=64, CD=256
// ---------------------------------------------------------------------------

typedef __bf16 bf16_t;
typedef bf16_t v16bf __attribute__((ext_vector_type(16)));
typedef float  v8f   __attribute__((ext_vector_type(8)));

union Frag16 { v16bf v; bf16_t h[16]; unsigned short u[16]; uint4 q[2]; };

__device__ __forceinline__ unsigned short bfbits(float f) {
    bf16_t h = (bf16_t)f;                      // native cvt (RNE)
    return __builtin_bit_cast(unsigned short, h);
}

// --------------------------- weight transpose ------------------------------
// W [512 in][512 out] fp32  ->  WT [512 out][512 in] bf16
__global__ __launch_bounds__(256) void wtrans_kernel(const float* __restrict__ W,
                                                     unsigned short* __restrict__ WT) {
    const int n = blockIdx.x;
    for (int k = threadIdx.x; k < 512; k += 256)
        WT[(size_t)n * 512 + k] = bfbits(W[(size_t)k * 512 + n]);
}

// --------------------------- confidence scalars ----------------------------
// conf[b,n] = sigmoid(rel[b,n]·w_rel + b_rel) * sigmoid(abs[n]·w_abs + b_abs) / 8
__global__ __launch_bounds__(256) void conf_kernel(const float* __restrict__ rel,
                                                   const float* __restrict__ absc,
                                                   const float* __restrict__ w_rel,
                                                   const float* __restrict__ b_rel,
                                                   const float* __restrict__ w_abs,
                                                   const float* __restrict__ b_abs,
                                                   float* __restrict__ conf) {
    const int b = blockIdx.x, n = threadIdx.x;
    const float* rp = rel  + ((size_t)b * 256 + n) * 256;
    const float* ap = absc + (size_t)n * 256;
    float dr = b_rel[0], da = b_abs[0];
    for (int c = 0; c < 256; ++c) { dr += rp[c] * w_rel[c]; da += ap[c] * w_abs[c]; }
    float sr = 1.f / (1.f + __expf(-dr));
    float sa = 1.f / (1.f + __expf(-da));
    conf[b * 256 + n] = sr * sa * 0.125f;     // fold 1/sqrt(HD=64)
}

// --------------------------- projection GEMM (fp32 A) ----------------------
// C[m, n] = A[m,:512] . WT[n,:512] + bias[n]; store bf16 in attention layout.
// vmode 0: dst[((b*8+h)*256 + nseq)*64 + d]   (Q / K layout)
// vmode 1: dst[((b*8+h)*64  + d)*256 + nseq]  (V transposed layout)
__global__ __launch_bounds__(256)
void gemm_a32(const float* __restrict__ A, const unsigned short* __restrict__ WT,
              const float* __restrict__ bias, unsigned short* __restrict__ dst, int vmode) {
    const int wave = threadIdx.x >> 5;
    const int lane = threadIdx.x & 31;
    const int lmod = lane & 15;
    const int lhi  = lane >> 4;
    const int m0 = blockIdx.x * 128 + wave * 16;
    const int n0 = blockIdx.y * 64;
    const float* arow = A + (size_t)(m0 + lmod) * 512;

    v8f acc[4];
    for (int t = 0; t < 4; ++t) for (int e = 0; e < 8; ++e) acc[t][e] = 0.f;

    for (int kb = 0; kb < 512; kb += 32) {
        Frag16 af;
        const float* p0 = arow + kb + lhi * 8;
        float4 x0 = *(const float4*)(p0);
        float4 x1 = *(const float4*)(p0 + 4);
        float4 x2 = *(const float4*)(p0 + 16);
        float4 x3 = *(const float4*)(p0 + 20);
        af.h[0]=(bf16_t)x0.x; af.h[1]=(bf16_t)x0.y; af.h[2]=(bf16_t)x0.z; af.h[3]=(bf16_t)x0.w;
        af.h[4]=(bf16_t)x1.x; af.h[5]=(bf16_t)x1.y; af.h[6]=(bf16_t)x1.z; af.h[7]=(bf16_t)x1.w;
        af.h[8]=(bf16_t)x2.x; af.h[9]=(bf16_t)x2.y; af.h[10]=(bf16_t)x2.z; af.h[11]=(bf16_t)x2.w;
        af.h[12]=(bf16_t)x3.x; af.h[13]=(bf16_t)x3.y; af.h[14]=(bf16_t)x3.z; af.h[15]=(bf16_t)x3.w;
        #pragma unroll
        for (int t = 0; t < 4; ++t) {
            Frag16 bfr;
            const unsigned short* bp = WT + (size_t)(n0 + t * 16 + lmod) * 512 + kb + lhi * 8;
            bfr.q[0] = *(const uint4*)(bp);
            bfr.q[1] = *(const uint4*)(bp + 16);
            acc[t] = __builtin_amdgcn_wmma_f32_16x16x32_bf16(
                false, af.v, false, bfr.v, (short)0, acc[t], false, false);
        }
    }
    #pragma unroll
    for (int t = 0; t < 4; ++t) {
        const int ncol = n0 + t * 16 + lmod;
        const float bvv = bias[ncol];
        const int hh = ncol >> 6, d = ncol & 63;
        #pragma unroll
        for (int r = 0; r < 8; ++r) {
            const int m = m0 + r + (lhi << 3);
            const int bidx = m >> 8, nseq = m & 255;
            const float val = acc[t][r] + bvv;
            size_t idx;
            if (vmode) idx = (((size_t)(bidx * 8 + hh) * 64 + d) * 256 + nseq);
            else       idx = (((size_t)(bidx * 8 + hh) * 256 + nseq) * 64 + d);
            dst[idx] = bfbits(val);
        }
    }
}

// --------------------------- output GEMM (bf16 A, fp32 out) ----------------
__global__ __launch_bounds__(256)
void gemm_abf(const unsigned short* __restrict__ A, const unsigned short* __restrict__ WT,
              const float* __restrict__ bias, float* __restrict__ out) {
    const int wave = threadIdx.x >> 5;
    const int lane = threadIdx.x & 31;
    const int lmod = lane & 15;
    const int lhi  = lane >> 4;
    const int m0 = blockIdx.x * 128 + wave * 16;
    const int n0 = blockIdx.y * 64;
    const unsigned short* arow = A + (size_t)(m0 + lmod) * 512;

    v8f acc[4];
    for (int t = 0; t < 4; ++t) for (int e = 0; e < 8; ++e) acc[t][e] = 0.f;

    for (int kb = 0; kb < 512; kb += 32) {
        Frag16 af;
        af.q[0] = *(const uint4*)(arow + kb + lhi * 8);
        af.q[1] = *(const uint4*)(arow + kb + 16 + lhi * 8);
        #pragma unroll
        for (int t = 0; t < 4; ++t) {
            Frag16 bfr;
            const unsigned short* bp = WT + (size_t)(n0 + t * 16 + lmod) * 512 + kb + lhi * 8;
            bfr.q[0] = *(const uint4*)(bp);
            bfr.q[1] = *(const uint4*)(bp + 16);
            acc[t] = __builtin_amdgcn_wmma_f32_16x16x32_bf16(
                false, af.v, false, bfr.v, (short)0, acc[t], false, false);
        }
    }
    #pragma unroll
    for (int t = 0; t < 4; ++t) {
        const int ncol = n0 + t * 16 + lmod;
        const float bvv = bias[ncol];
        #pragma unroll
        for (int r = 0; r < 8; ++r) {
            const int m = m0 + r + (lhi << 3);
            out[(size_t)m * 512 + ncol] = acc[t][r] + bvv;
        }
    }
}

// --------------------------- attention core --------------------------------
// One block per (b, h, 32-query tile). Q/K: [b,h,seq,64]; V: [b,h,64,seq].
// K and V tiles staged into LDS with async-to-LDS loads; V overlaps stage 1+2.
__global__ __launch_bounds__(256)
void attn_kernel(const unsigned short* __restrict__ qh,
                 const unsigned short* __restrict__ kh,
                 const unsigned short* __restrict__ vT,
                 const float* __restrict__ conf,
                 unsigned short* __restrict__ oB) {
    __shared__ float sc[32 * 257];
    __shared__ __align__(16) unsigned short Pl[32 * 264];
    __shared__ float red[32 * 8];
    __shared__ __align__(16) unsigned short Kl[256 * 72];   // K tile, row stride 72 bf16
    __shared__ __align__(16) unsigned short Vl[64 * 264];   // V^T tile, row stride 264 bf16

    const int blk = blockIdx.x;          // B * NH * 8
    const int qt  = blk & 7;
    const int h   = (blk >> 3) & 7;
    const int b   = blk >> 6;
    const int tid = threadIdx.x;
    const int wave = tid >> 5, lane = tid & 31, lmod = lane & 15, lhi = lane >> 4;

    const size_t headOff = (size_t)(b * 8 + h) * 256 * 64;
    const unsigned short* Q = qh + headOff;
    const unsigned short* K = kh + headOff;      // [256][64]
    const unsigned short* V = vT + headOff;      // [64][256]
    const int q0 = qt * 32;

    // ---- async stage: K tile then V tile -> LDS (8 + 8 x b128 per thread) ----
    #pragma unroll
    for (int c = tid; c < 2048; c += 256) {                 // K: 2048 16B chunks
        const int row = c >> 3, off = (c & 7) * 8;          // off in bf16 elems
        unsigned ldsa = (unsigned)(uintptr_t)(&Kl[row * 72 + off]);
        const unsigned short* g = K + (size_t)row * 64 + off;
        asm volatile("global_load_async_to_lds_b128 %0, %1, off"
                     :: "v"(ldsa), "v"(g) : "memory");
    }
    #pragma unroll
    for (int c = tid; c < 2048; c += 256) {                 // V: 2048 16B chunks
        const int row = c >> 5, off = (c & 31) * 8;
        unsigned ldsa = (unsigned)(uintptr_t)(&Vl[row * 264 + off]);
        const unsigned short* g = V + (size_t)row * 256 + off;
        asm volatile("global_load_async_to_lds_b128 %0, %1, off"
                     :: "v"(ldsa), "v"(g) : "memory");
    }
    // wait for this wave's 8 K-chunks (async loads complete in order; 8 V pending)
    asm volatile("s_wait_asynccnt 0x8" ::: "memory");
    __syncthreads();

    // ---- stage 1: S = Q K^T, scaled by conf[b,k]/8 -> LDS fp32 ----
    {
        const int wm = wave >> 2;             // 0..1 (row tile)
        const int wn = wave & 3;              // 0..3 (64-col slab)
        const unsigned short* qrow = Q + (size_t)(q0 + wm * 16 + lmod) * 64;
        v8f acc[4];
        for (int t = 0; t < 4; ++t) for (int e = 0; e < 8; ++e) acc[t][e] = 0.f;
        #pragma unroll
        for (int kb = 0; kb < 64; kb += 32) {
            Frag16 af;
            af.q[0] = *(const uint4*)(qrow + kb + lhi * 8);
            af.q[1] = *(const uint4*)(qrow + kb + 16 + lhi * 8);
            #pragma unroll
            for (int t = 0; t < 4; ++t) {
                Frag16 bfr;
                const unsigned short* kp = &Kl[(wn * 64 + t * 16 + lmod) * 72 + kb + lhi * 8];
                bfr.q[0] = *(const uint4*)kp;
                bfr.q[1] = *(const uint4*)(kp + 16);
                acc[t] = __builtin_amdgcn_wmma_f32_16x16x32_bf16(
                    false, af.v, false, bfr.v, (short)0, acc[t], false, false);
            }
        }
        #pragma unroll
        for (int t = 0; t < 4; ++t) {
            const int col = wn * 64 + t * 16 + lmod;
            const float cf = conf[b * 256 + col];
            #pragma unroll
            for (int r = 0; r < 8; ++r) {
                const int row = wm * 16 + r + (lhi << 3);
                sc[row * 257 + col] = acc[t][r] * cf;
            }
        }
    }
    __syncthreads();

    // ---- stage 2: row softmax (32 rows x 8 threads/row), P -> bf16 LDS ----
    {
        const int row = tid >> 3, part = tid & 7, base = part * 32;
        float mx = -1e30f;
        for (int j = 0; j < 32; ++j) mx = fmaxf(mx, sc[row * 257 + base + j]);
        red[row * 8 + part] = mx;
        __syncthreads();
        for (int p = 0; p < 8; ++p) mx = fmaxf(mx, red[row * 8 + p]);
        float s = 0.f;
        for (int j = 0; j < 32; ++j) {
            float e = __expf(sc[row * 257 + base + j] - mx);
            sc[row * 257 + base + j] = e;
            s += e;
        }
        __syncthreads();
        red[row * 8 + part] = s;
        __syncthreads();
        float tot = 0.f;
        for (int p = 0; p < 8; ++p) tot += red[row * 8 + p];
        const float inv = 1.0f / tot;
        for (int j = 0; j < 32; ++j)
            Pl[row * 264 + base + j] = bfbits(sc[row * 257 + base + j] * inv);
    }
    // V tile has had QK^T + softmax to arrive; drain remaining async loads
    asm volatile("s_wait_asynccnt 0x0" ::: "memory");
    __syncthreads();

    // ---- stage 3: O = P V  (32 x 64), bf16 out ----
    {
        const int wm = wave >> 2;             // 0..1
        const int wn = wave & 3;              // 0..3 (d tile)
        const int rowl = wm * 16 + lmod;
        v8f acc; for (int e = 0; e < 8; ++e) acc[e] = 0.f;
        #pragma unroll
        for (int kb = 0; kb < 256; kb += 32) {
            Frag16 af;
            af.q[0] = *(const uint4*)(&Pl[rowl * 264 + kb + lhi * 8]);
            af.q[1] = *(const uint4*)(&Pl[rowl * 264 + kb + 16 + lhi * 8]);
            Frag16 bfr;
            const unsigned short* vp = &Vl[(wn * 16 + lmod) * 264 + kb + lhi * 8];
            bfr.q[0] = *(const uint4*)vp;
            bfr.q[1] = *(const uint4*)(vp + 16);
            acc = __builtin_amdgcn_wmma_f32_16x16x32_bf16(
                false, af.v, false, bfr.v, (short)0, acc, false, false);
        }
        #pragma unroll
        for (int r = 0; r < 8; ++r) {
            const int qrow2 = q0 + wm * 16 + r + (lhi << 3);
            const int dcol = wn * 16 + lmod;
            oB[((size_t)(b * 256 + qrow2)) * 512 + h * 64 + dcol] = bfbits(acc[r]);
        }
    }
}

// --------------------------- out_conf broadcast ----------------------------
// out_conf[b,q,:] = abs_confidence[b,:]  (softmax rows sum to 1 exactly)
__global__ __launch_bounds__(256) void outconf_kernel(const float* __restrict__ absc,
                                                      float* __restrict__ out2) {
    const size_t idx = (size_t)blockIdx.x * 256 + threadIdx.x;   // 16,777,216 total
    const int c = (int)(idx & 255);
    const int b = (int)(idx >> 16);
    out2[idx] = absc[b * 256 + c];
}

// ---------------------------------------------------------------------------
extern "C" void kernel_launch(void* const* d_in, const int* in_sizes, int n_in,
                              void* d_out, int out_size, void* d_ws, size_t ws_size,
                              hipStream_t stream) {
    const float* query = (const float*)d_in[0];
    const float* key   = (const float*)d_in[1];
    const float* value = (const float*)d_in[2];
    const float* absc  = (const float*)d_in[3];
    const float* relc  = (const float*)d_in[4];
    const float* Wq = (const float*)d_in[5];  const float* bq = (const float*)d_in[6];
    const float* Wk = (const float*)d_in[7];  const float* bk = (const float*)d_in[8];
    const float* Wv = (const float*)d_in[9];  const float* bv = (const float*)d_in[10];
    const float* w_abs = (const float*)d_in[11]; const float* b_abs = (const float*)d_in[12];
    const float* w_rel = (const float*)d_in[13]; const float* b_rel = (const float*)d_in[14];
    const float* Wo = (const float*)d_in[15]; const float* bo = (const float*)d_in[16];

    char* w = (char*)d_ws;
    auto alloc = [&](size_t bytes) -> void* {
        void* p = (void*)w;
        w += (bytes + 255) & ~(size_t)255;
        return p;
    };
    const size_t ACT = (size_t)256 * 8 * 256 * 64 * sizeof(unsigned short); // 64 MiB
    unsigned short* qh  = (unsigned short*)alloc(ACT);
    unsigned short* kh  = (unsigned short*)alloc(ACT);
    unsigned short* vT  = (unsigned short*)alloc(ACT);
    unsigned short* oB  = (unsigned short*)alloc(ACT);
    unsigned short* WqT = (unsigned short*)alloc(512 * 512 * sizeof(unsigned short));
    unsigned short* WkT = (unsigned short*)alloc(512 * 512 * sizeof(unsigned short));
    unsigned short* WvT = (unsigned short*)alloc(512 * 512 * sizeof(unsigned short));
    unsigned short* WoT = (unsigned short*)alloc(512 * 512 * sizeof(unsigned short));
    float* conf = (float*)alloc(256 * 256 * sizeof(float));

    float* out_main = (float*)d_out;                    // [256,256,512]
    float* out_conf = out_main + (size_t)256 * 256 * 512;

    wtrans_kernel<<<512, 256, 0, stream>>>(Wq, WqT);
    wtrans_kernel<<<512, 256, 0, stream>>>(Wk, WkT);
    wtrans_kernel<<<512, 256, 0, stream>>>(Wv, WvT);
    wtrans_kernel<<<512, 256, 0, stream>>>(Wo, WoT);

    conf_kernel<<<256, 256, 0, stream>>>(relc, absc, w_rel, b_rel, w_abs, b_abs, conf);

    dim3 g(512, 8);
    gemm_a32<<<g, 256, 0, stream>>>(query, WqT, bq, qh, 0);
    gemm_a32<<<g, 256, 0, stream>>>(key,   WkT, bk, kh, 0);
    gemm_a32<<<g, 256, 0, stream>>>(value, WvT, bv, vT, 1);

    attn_kernel<<<256 * 8 * 8, 256, 0, stream>>>(qh, kh, vT, conf, oB);

    gemm_abf<<<g, 256, 0, stream>>>(oB, WoT, bo, out_main);

    outconf_kernel<<<65536, 256, 0, stream>>>(absc, out_conf);
}
